// GraphInferencer_71545565217157
// MI455X (gfx1250) — matrix operand order repaired and checked by reading, not compiled, not run on hardware
//
#include <hip/hip_runtime.h>
#include <hip/hip_bf16.h>

// Problem constants (match reference)
constexpr int kN = 50000;   // nodes
constexpr int kE = 250000;  // edges (before self loops)
constexpr int kB = 128;     // graphs
constexpr int kD = 302;     // feature dim
constexpr int kQ = 600;     // query dim
constexpr int kC = 2000;    // classes
constexpr float kNegSlope = 0.2f;
constexpr float kInvSqrtQ = 0.04082482904638630163f; // 1/sqrt(600)

// padded K sizes (multiple of 32)
constexpr int kDp = 320;    // 302 -> 320
constexpr int kQp = 608;    // 600 -> 608
// padded N sizes (multiple of 16)
constexpr int kDn = 304;    // 302 -> 304 output columns

typedef __attribute__((ext_vector_type(16))) __bf16 v16bf;
typedef __attribute__((ext_vector_type(8)))  __bf16 v8bf;
typedef __attribute__((ext_vector_type(8)))  float  v8f;

// ---------- helpers ----------
__device__ __forceinline__ unsigned fenc(float f) {
  unsigned u = __float_as_uint(f);
  return (u & 0x80000000u) ? ~u : (u | 0x80000000u);
}
__device__ __forceinline__ float fdec(unsigned u) {
  u = (u & 0x80000000u) ? (u & 0x7FFFFFFFu) : ~u;
  return __uint_as_float(u);
}

__global__ void fill_f32(float* p, float v, int n) {
  int i = blockIdx.x * blockDim.x + threadIdx.x;
  if (i < n) p[i] = v;
}
__global__ void fill_u32(unsigned* p, unsigned v, int n) {
  int i = blockIdx.x * blockDim.x + threadIdx.x;
  if (i < n) p[i] = v;
}

// ---------- operand packing: f32 row-major -> zero-padded bf16 [rowsPad, Kp] ----------
__global__ void pack_rows_bf16(const float* __restrict__ src, __bf16* __restrict__ dst,
                               int rowsPad, int rowsReal, int K, int Kp)
{
  long long i = (long long)blockIdx.x * blockDim.x + threadIdx.x;
  if (i >= (long long)rowsPad * Kp) return;
  int r = (int)(i / Kp);
  int k = (int)(i % Kp);
  float v = (r < rowsReal && k < K) ? src[(size_t)r * K + k] : 0.f;
  dst[i] = (__bf16)v;
}

// f32 [K, N] -> bf16 [Np, Kp] (transposed pack, so GEMM always reads K-major columns)
__global__ void pack_cols_bf16(const float* __restrict__ src, __bf16* __restrict__ dst,
                               int Np, int Nreal, int K, int Kp)
{
  long long i = (long long)blockIdx.x * blockDim.x + threadIdx.x;
  if (i >= (long long)Np * Kp) return;
  int n = (int)(i / Kp);
  int k = (int)(i % Kp);
  float v = (n < Nreal && k < K) ? src[(size_t)k * Nreal + n] : 0.f;
  dst[i] = (__bf16)v;
}

// ---------- WMMA GEMM on packed bf16 operands ----------
// C[M,Nc](f32) = A[M,Kp](bf16) @ Bp[Ncp,Kp](bf16)^T  (+bias)
// One wave -> one 16x16 C tile; no bounds checks in the K loop.
__global__ __launch_bounds__(128)
void gemm_wmma_packed(const __bf16* __restrict__ A, const __bf16* __restrict__ Bp,
                      float* __restrict__ C, const float* __restrict__ bias,
                      int M, int Kp, int Nc)
{
  const int lane = threadIdx.x & 31;
  const int kh   = lane >> 4;     // wave half
  const int mr   = lane & 15;
  const int m0   = blockIdx.x * 16;
  const int tileN = blockIdx.y * blockDim.y + threadIdx.y;
  const int n0   = tileN * 16;
  if (n0 >= Nc) return;           // uniform per wave (Bp rows padded to 16)

  // A 16x32 bf16 lane layout: this lane supplies row (m0+mr),
  // K chunk [k0 + 16*kh, +16)
  const __bf16* aptr = A + (size_t)(m0 + mr) * Kp + kh * 16;
  // B 32x16 bf16 lane layout: this lane supplies column (n0+mr),
  // K chunk [k0 + 16*kh, +16)
  const __bf16* bptr = Bp + (size_t)(n0 + mr) * Kp + kh * 16;

  v8f acc = {0.f, 0.f, 0.f, 0.f, 0.f, 0.f, 0.f, 0.f};

  for (int k0 = 0; k0 < Kp; k0 += 32) {
    v8bf a0 = *(const v8bf*)(aptr);
    v8bf a1 = *(const v8bf*)(aptr + 8);
    v16bf av = __builtin_shufflevector(a0, a1, 0, 1, 2, 3, 4, 5, 6, 7,
                                       8, 9, 10, 11, 12, 13, 14, 15);
    v16bf bv = *(const v16bf*)(bptr);
    acc = __builtin_amdgcn_wmma_f32_16x16x32_bf16(false, av, false, bv,
                                                  (short)0, acc, false, false);
    aptr += 32;
    bptr += 32;
  }

  const int cn = n0 + mr;
#pragma unroll
  for (int r = 0; r < 8; ++r) {
    int cm = m0 + r + kh * 8;
    if (cm < M && cn < Nc) {
      float v = acc[r];
      if (bias) v += bias[cn];
      C[(size_t)cm * Nc + cn] = v;
    }
  }
}

// ---------- per-node dot products: a_s = h.att_src, a_d = h.att_dst (wave per node) ----------
__global__ void node_dots(const float* __restrict__ h, const float* __restrict__ as_w,
                          const float* __restrict__ ad_w, float* __restrict__ a_s,
                          float* __restrict__ a_d, int n, int d)
{
  int wv   = (blockIdx.x * blockDim.x + threadIdx.x) >> 5;
  int lane = threadIdx.x & 31;
  if (wv >= n) return;
  const float* row = h + (size_t)wv * d;
  float s0 = 0.f, s1 = 0.f;
  for (int k = lane; k < d; k += 32) {
    float v = row[k];
    s0 += v * as_w[k];
    s1 += v * ad_w[k];
  }
  for (int o = 16; o > 0; o >>= 1) {
    s0 += __shfl_down(s0, o, 32);
    s1 += __shfl_down(s1, o, 32);
  }
  if (lane == 0) { a_s[wv] = s0; a_d[wv] = s1; }
}

__device__ __forceinline__ void edge_sd(const int* __restrict__ ed, int e, int& s, int& d) {
  if (e < kE) { s = ed[e]; d = ed[kE + e]; }
  else        { s = e - kE; d = e - kE; }
}
__device__ __forceinline__ float edge_score(const float* a_s, const float* a_d, int s, int d) {
  float sc = a_s[s] + a_d[d];
  return sc > 0.f ? sc : kNegSlope * sc;
}

// ---------- segment softmax over dst (3 passes, atomics) ----------
__global__ void edge_max_k(const int* __restrict__ ed, const float* __restrict__ a_s,
                           const float* __restrict__ a_d, unsigned* __restrict__ smax, int en)
{
  int e = blockIdx.x * blockDim.x + threadIdx.x;
  if (e >= en) return;
  int s, d; edge_sd(ed, e, s, d);
  atomicMax(&smax[d], fenc(edge_score(a_s, a_d, s, d)));
}

__global__ void edge_expsum_k(const int* __restrict__ ed, const float* __restrict__ a_s,
                              const float* __restrict__ a_d, const unsigned* __restrict__ smax,
                              float* __restrict__ ssum, int en)
{
  int e = blockIdx.x * blockDim.x + threadIdx.x;
  if (e >= en) return;
  int s, d; edge_sd(ed, e, s, d);
  float ex = expf(edge_score(a_s, a_d, s, d) - fdec(smax[d]));
  atomicAdd(&ssum[d], ex);
}

// wave per edge: agg[dst] += alpha * h[src]
__global__ void edge_scatter_k(const int* __restrict__ ed, const float* __restrict__ a_s,
                               const float* __restrict__ a_d, const unsigned* __restrict__ smax,
                               const float* __restrict__ ssum, const float* __restrict__ h,
                               float* __restrict__ agg, int en, int dim)
{
  int e    = (blockIdx.x * blockDim.x + threadIdx.x) >> 5;
  int lane = threadIdx.x & 31;
  if (e >= en) return;
  int s, d; edge_sd(ed, e, s, d);
  float alpha = expf(edge_score(a_s, a_d, s, d) - fdec(smax[d])) / ssum[d];
  const float* hr = h + (size_t)s * dim;
  float*       ar = agg + (size_t)d * dim;
  for (int k = lane; k < dim; k += 32)
    atomicAdd(&ar[k], alpha * hr[k]);
}

__global__ void relu_bias_k(float* __restrict__ h, const float* __restrict__ bias, int n, int d) {
  long long i = (long long)blockIdx.x * blockDim.x + threadIdx.x;
  if (i >= (long long)n * d) return;
  int col = (int)(i % d);
  float v = h[i] + bias[col];
  h[i] = v > 0.f ? v : 0.f;
}

// ---------- pooling ----------
__global__ void pool_score_k(const float* __restrict__ qp, const int* __restrict__ batch,
                             const float* __restrict__ hf, float* __restrict__ sval,
                             unsigned* __restrict__ bmax, int n, int d)
{
  int wv   = (blockIdx.x * blockDim.x + threadIdx.x) >> 5;
  int lane = threadIdx.x & 31;
  if (wv >= n) return;
  int b = batch[wv];
  const float* qr = qp + (size_t)b * d;
  const float* hr = hf + (size_t)wv * d;
  float s0 = 0.f;
  for (int k = lane; k < d; k += 32) s0 += qr[k] * hr[k];
  for (int o = 16; o > 0; o >>= 1) s0 += __shfl_down(s0, o, 32);
  if (lane == 0) {
    float sc = s0 * kInvSqrtQ;
    sval[wv] = sc;
    atomicMax(&bmax[b], fenc(sc));
  }
}

__global__ void pool_expsum_k(const int* __restrict__ batch, float* __restrict__ sval,
                              const unsigned* __restrict__ bmax, float* __restrict__ bsum, int n)
{
  int i = blockIdx.x * blockDim.x + threadIdx.x;
  if (i >= n) return;
  int b = batch[i];
  float ex = expf(sval[i] - fdec(bmax[b]));
  sval[i] = ex;
  atomicAdd(&bsum[b], ex);
}

__global__ void pool_scatter_k(const int* __restrict__ batch, const float* __restrict__ sval,
                               const float* __restrict__ bsum, const float* __restrict__ hf,
                               float* __restrict__ pooled, int n, int d)
{
  int wv   = (blockIdx.x * blockDim.x + threadIdx.x) >> 5;
  int lane = threadIdx.x & 31;
  if (wv >= n) return;
  int b = batch[wv];
  float w = sval[wv] / bsum[b];
  const float* hr = hf + (size_t)wv * d;
  float*       pr = pooled + (size_t)b * d;
  for (int k = lane; k < d; k += 32)
    atomicAdd(&pr[k], w * hr[k]);
}

__global__ void relu_k(float* p, int n) {
  int i = blockIdx.x * blockDim.x + threadIdx.x;
  if (i < n) p[i] = p[i] > 0.f ? p[i] : 0.f;
}

// ---------- driver ----------
static inline int cdiv(int a, int b) { return (a + b - 1) / b; }

extern "C" void kernel_launch(void* const* d_in, const int* in_sizes, int n_in,
                              void* d_out, int out_size, void* d_ws, size_t ws_size,
                              hipStream_t stream)
{
  (void)in_sizes; (void)n_in; (void)out_size; (void)ws_size;
  const float* x        = (const float*)d_in[0];
  const int*   edges    = (const int*)d_in[1];
  const float* query    = (const float*)d_in[2];
  const int*   batch    = (const int*)d_in[3];
  const float* theta    = (const float*)d_in[4];
  const float* att_src  = (const float*)d_in[5];
  const float* att_dst  = (const float*)d_in[6];
  const float* gat_bias = (const float*)d_in[7];
  const float* attW     = (const float*)d_in[8];
  const float* lin_w    = (const float*)d_in[9];
  const float* lin_b    = (const float*)d_in[10];
  float* out = (float*)d_out;

  char* w = (char*)d_ws;
  auto take = [&](size_t bytes) { char* p = w; w += (bytes + 255) & ~(size_t)255; return p; };
  float*    h      = (float*)take((size_t)kN * kD * 4);
  float*    agg    = (float*)take((size_t)kN * kD * 4);
  __bf16*   Abf    = (__bf16*)take((size_t)kN * kDp * 2);   // packed activations
  __bf16*   Bth    = (__bf16*)take((size_t)kDn * kDp * 2);  // theta   [304, 320]
  __bf16*   Blin   = (__bf16*)take((size_t)kC * kDp * 2);   // lin_w   [2000, 320]
  __bf16*   Batt   = (__bf16*)take((size_t)kDn * kQp * 2);  // attW^T  [304, 608]
  __bf16*   Qbf    = (__bf16*)take((size_t)kB * kQp * 2);   // query   [128, 608]
  __bf16*   Pbf    = (__bf16*)take((size_t)kB * kDp * 2);   // pooled  [128, 320]
  float*    a_s    = (float*)take((size_t)kN * 4);
  float*    a_d    = (float*)take((size_t)kN * 4);
  unsigned* smax   = (unsigned*)take((size_t)kN * 4);
  float*    ssum   = (float*)take((size_t)kN * 4);
  float*    sval   = (float*)take((size_t)kN * 4);
  unsigned* bmax   = (unsigned*)take((size_t)kB * 4);
  float*    bsum   = (float*)take((size_t)kB * 4);
  float*    qp     = (float*)take((size_t)kB * kD * 4);
  float*    pooled = (float*)take((size_t)kB * kD * 4);

  const int EN = kE + kN;            // edges incl. self loops
  const dim3 gemmBlk(32, 4);         // 4 waves per block, wave = one 16x16 C tile
  const int ND = kN * kD;

  // ---- one-time weight packing (bf16, zero-padded, K-major columns) ----
  pack_rows_bf16<<<cdiv(kDn * kDp, 256), 256, 0, stream>>>(theta, Bth, kDn, kD, kD, kDp);
  pack_rows_bf16<<<cdiv(kC * kDp, 256), 256, 0, stream>>>(lin_w, Blin, kC, kC, kD, kDp);
  pack_cols_bf16<<<cdiv(kDn * kQp, 256), 256, 0, stream>>>(attW, Batt, kDn, kD, kQ, kQp);
  pack_rows_bf16<<<cdiv(kB * kQp, 256), 256, 0, stream>>>(query, Qbf, kB, kB, kQ, kQp);

  const float* layer_in = x;
  for (int L = 0; L < 3; ++L) {
    // pack activations, then h = layer_in @ theta^T  [kN, kD]
    pack_rows_bf16<<<cdiv(kN * kDp, 256), 256, 0, stream>>>(layer_in, Abf, kN, kN, kD, kDp);
    dim3 gg(kN / 16, cdiv(cdiv(kD, 16), 4));
    gemm_wmma_packed<<<gg, gemmBlk, 0, stream>>>(Abf, Bth, h, nullptr, kN, kDp, kD);
    // attention logits per node
    node_dots<<<cdiv(kN, 8), 256, 0, stream>>>(h, att_src, att_dst, a_s, a_d, kN, kD);
    // reset accumulators
    fill_f32<<<cdiv(ND, 256), 256, 0, stream>>>(agg, 0.f, ND);
    fill_u32<<<cdiv(kN, 256), 256, 0, stream>>>(smax, 0u, kN);
    fill_f32<<<cdiv(kN, 256), 256, 0, stream>>>(ssum, 0.f, kN);
    // segment softmax over dst + weighted scatter
    edge_max_k   <<<cdiv(EN, 256), 256, 0, stream>>>(edges, a_s, a_d, smax, EN);
    edge_expsum_k<<<cdiv(EN, 256), 256, 0, stream>>>(edges, a_s, a_d, smax, ssum, EN);
    edge_scatter_k<<<cdiv(EN, 8), 256, 0, stream>>>(edges, a_s, a_d, smax, ssum, h, agg, EN, kD);
    // relu(agg + bias) in place -> next layer input
    relu_bias_k<<<cdiv(ND, 256), 256, 0, stream>>>(agg, gat_bias, kN, kD);
    layer_in = agg;
  }

  // qp = query @ attW   [kB, kD]
  {
    dim3 gg(kB / 16, cdiv(cdiv(kD, 16), 4));
    gemm_wmma_packed<<<gg, gemmBlk, 0, stream>>>(Qbf, Batt, qp, nullptr, kB, kQp, kD);
  }
  // pooling softmax over batch segments
  fill_u32<<<1, 256, 0, stream>>>(bmax, 0u, kB);
  fill_f32<<<1, 256, 0, stream>>>(bsum, 0.f, kB);
  fill_f32<<<cdiv(kB * kD, 256), 256, 0, stream>>>(pooled, 0.f, kB * kD);
  pool_score_k  <<<cdiv(kN, 8), 256, 0, stream>>>(qp, batch, agg, sval, bmax, kN, kD);
  pool_expsum_k <<<cdiv(kN, 256), 256, 0, stream>>>(batch, sval, bmax, bsum, kN);
  pool_scatter_k<<<cdiv(kN, 8), 256, 0, stream>>>(batch, sval, bsum, agg, pooled, kN, kD);
  relu_k<<<cdiv(kB * kD, 256), 256, 0, stream>>>(pooled, kB * kD);

  // out = relu(pooled) @ lin_w^T + lin_b   [kB, kC]
  {
    pack_rows_bf16<<<cdiv(kB * kDp, 256), 256, 0, stream>>>(pooled, Pbf, kB, kB, kD, kDp);
    dim3 gg(kB / 16, cdiv(cdiv(kC, 16), 4));
    gemm_wmma_packed<<<gg, gemmBlk, 0, stream>>>(Pbf, Blin, out, lin_b, kB, kDp, kC);
  }
}
